// RPN_19593640804914
// MI455X (gfx1250) — matrix operand order repaired and checked by reading, not compiled, not run on hardware
//
#include <hip/hip_runtime.h>
#include <hip/hip_bf16.h>
#include <cstdint>

typedef __attribute__((ext_vector_type(16))) _Float16 v16h;
typedef __attribute__((ext_vector_type(8)))  float    v8f;
typedef __attribute__((ext_vector_type(4)))  unsigned u32x4;
typedef __attribute__((ext_vector_type(8)))  int      i32x8;
typedef __attribute__((ext_vector_type(4)))  int      i32x4;

// ---------------- problem constants ----------------
constexpr int Bc  = 2;      // batch
constexpr int Cc  = 1024;   // in channels
constexpr int Hc  = 38, Wc = 50;
constexpr int COc = 512;    // conv out channels
constexpr int Hp  = 40, Wp = 52;            // padded spatial (halo 1)
constexpr int Mm  = Hc * Wc;                // 1900 spatial positions
constexpr int MpD = 1904;                   // padded to /16
constexpr int Kc  = Cc * 9;                 // 9216
constexpr int ANC = 9;
constexpr int NAn = Mm * ANC;               // 17100 anchors
constexpr int PRE = 6000, POST = 300;
constexpr int LEN = 32768;                  // sort pad (pow2 >= 17100)
constexpr int Gc  = 20;                     // gt boxes

// ---------------- workspace layout (bytes) ----------------
constexpr size_t AL(size_t x) { return (x + 255) & ~(size_t)255; }
constexpr size_t OFF_WH  = 0;                                   // f16 [9216][512]
constexpr size_t OFF_FH  = AL(OFF_WH  + (size_t)Kc*COc*2);      // f16 [B][40][52][1024]
constexpr size_t OFF_XH  = AL(OFF_FH  + (size_t)Bc*Hp*Wp*Cc*2); // f16 [B][1904][512]
constexpr size_t OFF_WCR = AL(OFF_XH  + (size_t)Bc*MpD*COc*2);  // f16 [512][64]
constexpr size_t OFF_CLS = AL(OFF_WCR + (size_t)COc*64*2);      // f32 [B][1904][18]
constexpr size_t OFF_REG = AL(OFF_CLS + (size_t)Bc*MpD*18*4);   // f32 [B][1904][36]
constexpr size_t OFF_BOX = AL(OFF_REG + (size_t)Bc*MpD*36*4);   // f32 [B][17100][4]
constexpr size_t OFF_KEY = AL(OFF_BOX + (size_t)Bc*NAn*4*4);    // u64 [B][32768]
constexpr size_t OFF_TB  = AL(OFF_KEY + (size_t)Bc*LEN*8);      // f32 [B][6000][4]
constexpr size_t OFF_LAB = AL(OFF_TB  + (size_t)Bc*PRE*4*4);    // i32 [B][17100]
constexpr size_t OFF_AMX = AL(OFF_LAB + (size_t)Bc*NAn*4);      // i32 [B][17100]
constexpr size_t OFF_MOV = AL(OFF_AMX + (size_t)Bc*NAn*4);      // f32 [B][17100]
constexpr size_t OFF_GTM = AL(OFF_MOV + (size_t)Bc*NAn*4);      // f32 [B][20]
constexpr size_t OFF_TGT = AL(OFF_GTM + 256);                   // f32 [B][17100][4]
constexpr size_t OFF_NEX = AL(OFF_TGT + (size_t)Bc*NAn*4*4);    // f32 [B]
constexpr size_t OFF_ACC = AL(OFF_NEX + 256);                   // f32 [16]

// ---------------- device helpers ----------------
__device__ __forceinline__ void anchor_of(int n, float& x1, float& y1, float& x2, float& y2) {
  int m = n / ANC, a = n - m * ANC;
  int ri = a / 3, si = a - ri * 3;
  float ratio = (ri == 0) ? 0.5f : ((ri == 1) ? 1.0f : 2.0f);
  float scale = (si == 0) ? 8.0f : ((si == 1) ? 16.0f : 32.0f);
  float ws = roundf(sqrtf(256.0f / ratio));
  float hs = roundf(ws * ratio);
  float rx1 = 7.5f - 0.5f * (ws - 1.0f), ry1 = 7.5f - 0.5f * (hs - 1.0f);
  float rx2 = 7.5f + 0.5f * (ws - 1.0f), ry2 = 7.5f + 0.5f * (hs - 1.0f);
  float w = rx2 - rx1 + 1.0f, h = ry2 - ry1 + 1.0f;
  float cx = rx1 + 0.5f * (w - 1.0f), cy = ry1 + 0.5f * (h - 1.0f);
  float W2 = w * scale, H2 = h * scale;
  float sx = (float)((m % Wc) * 16), sy = (float)((m / Wc) * 16);
  x1 = sx + cx - 0.5f * (W2 - 1.0f); y1 = sy + cy - 0.5f * (H2 - 1.0f);
  x2 = sx + cx + 0.5f * (W2 - 1.0f); y2 = sy + cy + 0.5f * (H2 - 1.0f);
}

__device__ __forceinline__ float iou_f(float ax1, float ay1, float ax2, float ay2,
                                       const float* g) {
  float iw = fmaxf(fminf(ax2, g[2]) - fmaxf(ax1, g[0]) + 1.0f, 0.0f);
  float ih = fmaxf(fminf(ay2, g[3]) - fmaxf(ay1, g[1]) + 1.0f, 0.0f);
  float inter = iw * ih;
  float aa = (ax2 - ax1 + 1.0f) * (ay2 - ay1 + 1.0f);
  float ab = (g[2] - g[0] + 1.0f) * (g[3] - g[1] + 1.0f);
  return inter / (aa + ab - inter);
}

__device__ __forceinline__ unsigned int fkey(float f) {
  unsigned int u = __float_as_uint(f);
  return (u & 0x80000000u) ? ~u : (u | 0x80000000u);
}

// TDM: 2D tile load Global -> LDS. data_size=2 bytes, tile tile_d0 x tile_d1,
// row stride stride0 (elements). Descriptor per CDNA5 ISA ch.8 (D# groups 0/1),
// groups 2/3 zero (<=2D tensor), type=2 ("image"), count=1.
__device__ __forceinline__ void tdm_load_2d(unsigned lds_off, const void* gaddr,
                                            unsigned tile_d0, unsigned tile_d1,
                                            unsigned tensor_d0, unsigned tensor_d1,
                                            unsigned stride0) {
  unsigned long long ga = (unsigned long long)(uintptr_t)gaddr;
  u32x4 g0;
  g0.x = 1u;                                         // count=1 (valid), user mode
  g0.y = lds_off;                                    // lds_addr (bytes)
  g0.z = (unsigned)(ga & 0xFFFFFFFFull);             // global_addr[31:0]
  g0.w = (unsigned)((ga >> 32) & 0x1FFFFFFull) | (2u << 30); // ga[56:32], type=2
  i32x8 g1;
  g1[0] = (int)(1u << 16);                           // wg_mask=0, data_size=1 (2B)
  g1[1] = (int)((tensor_d0 & 0xFFFFu) << 16);        // tensor_dim0[15:0]
  g1[2] = (int)(((tensor_d0 >> 16) & 0xFFFFu) | ((tensor_d1 & 0xFFFFu) << 16));
  g1[3] = (int)(((tensor_d1 >> 16) & 0xFFFFu) | ((tile_d0 & 0xFFFFu) << 16));
  g1[4] = (int)(tile_d1 & 0xFFFFu);                  // tile_dim1; tile_dim2=0
  g1[5] = (int)stride0;                              // tensor_dim0_stride[31:0]
  g1[6] = 0;                                         // stride0[47:32], stride1[15:0]
  g1[7] = 0;
  i32x4 z4 = {0, 0, 0, 0};
#if defined(__clang_major__) && (__clang_major__ >= 23)
  i32x8 z8 = {0, 0, 0, 0, 0, 0, 0, 0};
  __builtin_amdgcn_tensor_load_to_lds(g0, g1, z4, z4, z8, 0);
#else
  __builtin_amdgcn_tensor_load_to_lds(g0, g1, z4, z4, 0);
#endif
}

// ---------------- packing kernels ----------------
__global__ void zero_acc_k(float* acc) { if (threadIdx.x < 16) acc[threadIdx.x] = 0.0f; }

// Wh[(r*1024+c)*512+co] = w_conv[co][c][r]  (f32 -> f16)
__global__ void pack_w_k(const float* __restrict__ wconv, _Float16* __restrict__ Wh) {
  int i = blockIdx.x * 256 + threadIdx.x;
  if (i >= Kc * COc) return;
  int co = i & 511, k = i >> 9;
  int r = k >> 10, c = k & 1023;
  Wh[i] = (_Float16)wconv[((size_t)co * Cc + c) * 9 + r];
}

// Fh[b][yp][xp][c] : NHWC, zero halo
__global__ void pack_f_k(const float* __restrict__ f, _Float16* __restrict__ Fh) {
  int i = blockIdx.x * 256 + threadIdx.x;
  if (i >= Bc * Hp * Wp * Cc) return;
  int c = i & 1023;
  int t = i >> 10;
  int xp = t % Wp; t /= Wp;
  int yp = t % Hp; int b = t / Hp;
  int iy = yp - 1, ix = xp - 1;
  float v = 0.0f;
  if (iy >= 0 && iy < Hc && ix >= 0 && ix < Wc)
    v = f[(((size_t)b * Cc + c) * Hc + iy) * Wc + ix];
  Fh[i] = (_Float16)v;
}

// Wcr[k][n] : n<18 cls, 18<=n<54 reg, else 0
__global__ void pack_wcr_k(const float* __restrict__ wcls, const float* __restrict__ wreg,
                           _Float16* __restrict__ Wcr) {
  int i = blockIdx.x * 256 + threadIdx.x;
  if (i >= COc * 64) return;
  int n = i & 63, k = i >> 6;
  float v = 0.0f;
  if (n < 18) v = wcls[(size_t)n * COc + k];
  else if (n < 54) v = wreg[(size_t)(n - 18) * COc + k];
  Wcr[i] = (_Float16)v;
}

// ---------------- 3x3 conv: implicit GEMM, double-buffered TDM weights ------
// grid (30, 16, B), 256 threads = 8 waves: wave -> (wm 0..3, wn 0..1).
// Weights for this block's 32 output channels are DMA'd by the Tensor Data
// Mover in 36 chunks (256K x 32N f16 = 16 KB), double-buffered in LDS so the
// DMA of chunk k+1 overlaps the WMMA work on chunk k.
__global__ __launch_bounds__(256) void conv3_wmma_k(
    const _Float16* __restrict__ Fh, const _Float16* __restrict__ Wh,
    const float* __restrict__ bconv, _Float16* __restrict__ Xh) {
  __shared__ _Float16 smB[2][256 * 32];   // 2 x 16 KB, LDS offsets 0 / 16384
  int b = blockIdx.z;
  int wave = threadIdx.x >> 5, lane = threadIdx.x & 31;
  int m0 = blockIdx.x * 64 + (wave >> 1) * 16;
  int nblk = blockIdx.y * 32;          // block's N base
  int n0loc = (wave & 1) * 16;         // wave's N sub-tile inside LDS
  int row = lane & 15;
  int m = m0 + row; if (m > Mm - 1) m = Mm - 1;
  int y = m / Wc, x = m - y * Wc;
  int segoff = (lane < 16) ? 0 : 8;    // ISA 16-bit A layout: K {0..7,16..23}/{8..15,24..31}
  // Opaque-guarded store: the real writer of smB is the TDM engine, which
  // alias analysis cannot see. Without a visible may-store, LLVM folds the
  // ds_loads below to undef (observed in round 2). Never taken at runtime.
  if (__float_as_uint(bconv[0]) == 0x7F123456u) smB[0][threadIdx.x] = (_Float16)0.0f;
  v8f acc = {};
  if (wave == 0) {                     // prologue DMA: chunk 0 -> buffer 0
    tdm_load_2d(0u, Wh + nblk, 32u, 256u, 32u, 256u, (unsigned)COc);
  }
  for (int kidx = 0; kidx < 36; ++kidx) {      // 9 taps x 4 K-chunks
    if (wave == 0) __builtin_amdgcn_s_wait_tensorcnt(0);
    __syncthreads();                           // publish buf[kidx&1]
    if (wave == 0 && kidx + 1 < 36) {          // overlap: DMA next chunk
      int nr9 = (kidx + 1) >> 2, nch = (kidx + 1) & 3;
      const _Float16* gsrc = Wh + ((size_t)(nr9 * Cc + nch * 256)) * COc + nblk;
      tdm_load_2d((unsigned)(((kidx + 1) & 1) * 256 * 32 * 2), gsrc,
                  32u, 256u, 32u, 256u, (unsigned)COc);
    }
    int r9 = kidx >> 2, chk = kidx & 3;
    int ky = r9 / 3, kx = r9 - ky * 3;
    const _Float16* abase =
        Fh + (((size_t)(b * Hp + y + ky) * Wp) + x + kx) * Cc + chk * 256 + segoff;
    __builtin_prefetch(abase, 0, 1);           // global_prefetch_b8
    const _Float16* bbuf = &smB[kidx & 1][0];
    for (int kk = 0; kk < 256; kk += 32) {
      v16h af, bf;
      ((float4*)&af)[0] = *(const float4*)(abase + kk);
      ((float4*)&af)[1] = *(const float4*)(abase + kk + 16);
      for (int s = 0; s < 16; ++s)
        bf[s] = bbuf[(kk + segoff + (s & 7) + ((s >> 3) << 4)) * 32 + n0loc + (lane & 15)];
      acc = __builtin_amdgcn_wmma_f32_16x16x32_f16(
          false, af, false, bf, (short)0, acc, false, false);
    }
  }
  int co = nblk + (lane & 15);
  float bias = bconv[co];
  for (int rr = 0; rr < 8; ++rr) {
    int mm = m0 + ((lane < 16) ? rr : rr + 8);
    if (mm < Mm) {
      float v = fmaxf(acc[rr] + bias, 0.0f);   // bias + ReLU
      Xh[((size_t)b * MpD + mm) * COc + co] = (_Float16)v;
    }
  }
}

// ---------------- fused 1x1 cls/reg conv via WMMA ----------------
__global__ __launch_bounds__(256) void conv1_wmma_k(
    const _Float16* __restrict__ Xh, const _Float16* __restrict__ Wcr,
    const float* __restrict__ bcls, const float* __restrict__ breg,
    float* __restrict__ CLS, float* __restrict__ REGo) {
  int b = blockIdx.z;
  int wave = threadIdx.x >> 5, lane = threadIdx.x & 31;
  int id = blockIdx.x * 8 + wave;
  int mt = id >> 2, nt = id & 3;
  if (mt >= MpD / 16) return;
  int m0 = mt * 16, n0 = nt * 16;
  int m = m0 + (lane & 15);
  int segoff = (lane < 16) ? 0 : 8;
  const _Float16* abase = Xh + ((size_t)b * MpD + m) * COc + segoff;
  v8f acc = {};
  for (int k0 = 0; k0 < COc; k0 += 32) {
    v16h af, bf;
    ((float4*)&af)[0] = *(const float4*)(abase + k0);
    ((float4*)&af)[1] = *(const float4*)(abase + k0 + 16);
    for (int s = 0; s < 16; ++s)
      bf[s] = Wcr[(size_t)(k0 + segoff + (s & 7) + ((s >> 3) << 4)) * 64 + n0 + (lane & 15)];
    acc = __builtin_amdgcn_wmma_f32_16x16x32_f16(
        false, af, false, bf, (short)0, acc, false, false);
  }
  int ch = n0 + (lane & 15);
  for (int rr = 0; rr < 8; ++rr) {
    int mm = m0 + ((lane < 16) ? rr : rr + 8);
    if (mm < Mm) {
      float v = acc[rr];
      if (ch < 18)      CLS[((size_t)b * MpD + mm) * 18 + ch] = v + bcls[ch];
      else if (ch < 54) REGo[((size_t)b * MpD + mm) * 36 + (ch - 18)] = v + breg[ch - 18];
    }
  }
}

// ---------------- proposal: boxes, scores, sort keys ----------------
__global__ void proposal_prep_k(const float* __restrict__ CLS, const float* __restrict__ REGo,
                                const float* __restrict__ info,
                                float* __restrict__ BOX, unsigned long long* __restrict__ KEY) {
  int i = blockIdx.x * 256 + threadIdx.x;
  if (i >= Bc * LEN) return;
  int b = i >> 15, n = i & (LEN - 1);
  unsigned long long key = 0ull;
  if (n < NAn) {
    int m = n / ANC, a = n - m * ANC;
    float ax1, ay1, ax2, ay2;
    anchor_of(n, ax1, ay1, ax2, ay2);
    float c0 = CLS[((size_t)b * MpD + m) * 18 + a];
    float c1 = CLS[((size_t)b * MpD + m) * 18 + 9 + a];
    float sc = 1.0f / (1.0f + expf(c0 - c1));       // fg softmax prob
    const float* rg = REGo + ((size_t)b * MpD + m) * 36 + a * 4;
    float w = ax2 - ax1 + 1.0f, h = ay2 - ay1 + 1.0f;
    float cx = ax1 + 0.5f * w, cy = ay1 + 0.5f * h;
    float pcx = rg[0] * w + cx, pcy = rg[1] * h + cy;
    float pw = expf(rg[2]) * w, ph = expf(rg[3]) * h;
    float imh = info[b * 3] - 1.0f, imw = info[b * 3 + 1] - 1.0f;
    float x1 = fminf(fmaxf(pcx - 0.5f * pw, 0.0f), imw);
    float y1 = fminf(fmaxf(pcy - 0.5f * ph, 0.0f), imh);
    float x2 = fminf(fmaxf(pcx + 0.5f * pw, 0.0f), imw);
    float y2 = fminf(fmaxf(pcy + 0.5f * ph, 0.0f), imh);
    float* bx = BOX + ((size_t)b * NAn + n) * 4;
    bx[0] = x1; bx[1] = y1; bx[2] = x2; bx[3] = y2;
    float mins = 16.0f * info[b * 3 + 2];
    if (!((x2 - x1 + 1.0f >= mins) && (y2 - y1 + 1.0f >= mins))) sc = -1e9f;
    key = ((unsigned long long)fkey(sc) << 32) | (unsigned long long)(0xFFFFFFFFu - (unsigned)n);
  }
  KEY[(size_t)b * LEN + n] = key;
}

// descending bitonic pass over LEN keys per batch
__global__ void bitonic_pass_k(unsigned long long* __restrict__ KEY, int k, int j) {
  int i = blockIdx.x * 256 + threadIdx.x;
  int b = blockIdx.y;
  int ixj = i ^ j;
  if (ixj > i) {
    unsigned long long* kb = KEY + (size_t)b * LEN;
    unsigned long long a = kb[i], c = kb[ixj];
    bool descSeg = ((i & k) == 0);
    bool sw = descSeg ? (a < c) : (a > c);
    if (sw) { kb[i] = c; kb[ixj] = a; }
  }
}

__global__ void gather_top_k(const unsigned long long* __restrict__ KEY,
                             const float* __restrict__ BOX, float* __restrict__ TB) {
  int i = blockIdx.x * 256 + threadIdx.x;
  if (i >= Bc * PRE) return;
  int b = i / PRE, t = i - b * PRE;
  unsigned long long key = KEY[(size_t)b * LEN + t];
  unsigned idx = 0xFFFFFFFFu - (unsigned)(key & 0xFFFFFFFFull);
  if (idx >= (unsigned)NAn) idx = 0;
  const float* s = BOX + ((size_t)b * NAn + idx) * 4;
  float4 v = { s[0], s[1], s[2], s[3] };
  ((float4*)TB)[(size_t)b * PRE + t] = v;
}

// ---------------- sequential NMS (one block per batch) ----------------
__global__ __launch_bounds__(1024) void nms_k(const float* __restrict__ TB,
                                              float* __restrict__ out) {
  extern __shared__ char smem[];
  float4* sb = (float4*)smem;            // PRE boxes
  int* keep = (int*)(smem + PRE * 16);   // PRE flags
  int b = blockIdx.x, t = threadIdx.x;
  for (int i = t; i < PRE; i += 1024) {
    sb[i] = ((const float4*)TB)[(size_t)b * PRE + i];
    keep[i] = 1;
  }
  __syncthreads();
  for (int i = 0; i < PRE; ++i) {
    if (keep[i]) {
      float4 bi = sb[i];
      float areai = (bi.z - bi.x + 1.0f) * (bi.w - bi.y + 1.0f);
      for (int j = i + 1 + t; j < PRE; j += 1024) {
        if (keep[j]) {
          float4 bj = sb[j];
          float iw = fmaxf(fminf(bi.z, bj.z) - fmaxf(bi.x, bj.x) + 1.0f, 0.0f);
          float ih = fmaxf(fminf(bi.w, bj.w) - fmaxf(bi.y, bj.y) + 1.0f, 0.0f);
          float inter = iw * ih;
          float areaj = (bj.z - bj.x + 1.0f) * (bj.w - bj.y + 1.0f);
          if (inter / (areai + areaj - inter) > 0.7f) keep[j] = 0;
        }
      }
    }
    __syncthreads();
  }
  if (t == 0) {  // compact first POST kept (fill with index 0 like nonzero(fill=0))
    float* o = out + (size_t)b * POST * 5;
    int cnt = 0;
    for (int i = 0; i < PRE && cnt < POST; ++i) {
      if (keep[i]) {
        float4 bi = sb[i];
        o[cnt * 5 + 0] = (float)b; o[cnt * 5 + 1] = bi.x; o[cnt * 5 + 2] = bi.y;
        o[cnt * 5 + 3] = bi.z;     o[cnt * 5 + 4] = bi.w;
        cnt++;
      }
    }
    float4 f0 = sb[0];
    for (; cnt < POST; ++cnt) {
      o[cnt * 5 + 0] = (float)b; o[cnt * 5 + 1] = f0.x; o[cnt * 5 + 2] = f0.y;
      o[cnt * 5 + 3] = f0.z;     o[cnt * 5 + 4] = f0.w;
    }
  }
}

// ---------------- anchor targets ----------------
__global__ void maxov_k(const float* __restrict__ gt, const float* __restrict__ info,
                        float* __restrict__ MOV, int* __restrict__ AMX) {
  int i = blockIdx.x * 256 + threadIdx.x;
  if (i >= Bc * NAn) return;
  int b = i / NAn, n = i - b * NAn;
  float x1, y1, x2, y2; anchor_of(n, x1, y1, x2, y2);
  bool inside = (x1 >= 0.0f) && (y1 >= 0.0f) && (x2 < info[b * 3 + 1]) && (y2 < info[b * 3]);
  float mx = -2.0f; int am = 0;
  for (int g = 0; g < Gc; ++g) {
    const float* gp = gt + ((size_t)b * Gc + g) * 5;
    float ov = inside ? iou_f(x1, y1, x2, y2, gp) : -1.0f;
    if (ov > mx) { mx = ov; am = g; }
  }
  MOV[i] = mx; AMX[i] = am;
}

__global__ void gtmax_k(const float* __restrict__ gt, const float* __restrict__ info,
                        float* __restrict__ GTM) {
  int t = threadIdx.x;
  if (t >= Bc * Gc) return;
  int b = t / Gc, g = t - b * Gc;
  const float* gp = gt + ((size_t)b * Gc + g) * 5;
  float mx = -2.0f;
  for (int n = 0; n < NAn; ++n) {
    float x1, y1, x2, y2; anchor_of(n, x1, y1, x2, y2);
    bool inside = (x1 >= 0.0f) && (y1 >= 0.0f) && (x2 < info[b * 3 + 1]) && (y2 < info[b * 3]);
    float ov = inside ? iou_f(x1, y1, x2, y2, gp) : -1.0f;
    mx = fmaxf(mx, ov);
  }
  GTM[t] = mx;
}

__global__ void labels_k(const float* __restrict__ gt, const float* __restrict__ info,
                         const float* __restrict__ MOV, const float* __restrict__ GTM,
                         int* __restrict__ LAB) {
  int i = blockIdx.x * 256 + threadIdx.x;
  if (i >= Bc * NAn) return;
  int b = i / NAn, n = i - b * NAn;
  float x1, y1, x2, y2; anchor_of(n, x1, y1, x2, y2);
  bool inside = (x1 >= 0.0f) && (y1 >= 0.0f) && (x2 < info[b * 3 + 1]) && (y2 < info[b * 3]);
  int lab = -1;
  float mo = MOV[i];
  if (inside && mo < 0.3f) lab = 0;
  bool isg = false;
  for (int g = 0; g < Gc; ++g) {
    float gm = GTM[b * Gc + g];
    if (gm > 0.0f) {
      const float* gp = gt + ((size_t)b * Gc + g) * 5;
      float ov = inside ? iou_f(x1, y1, x2, y2, gp) : -1.0f;
      if (ov == gm) isg = true;
    }
  }
  if (inside && isg) lab = 1;
  if (inside && mo >= 0.7f) lab = 1;
  LAB[i] = lab;
}

__global__ void sample_k(int* __restrict__ LAB, float* __restrict__ NEX) {
  int b = threadIdx.x;
  if (b >= Bc) return;
  int* lab = LAB + (size_t)b * NAn;
  int fg = 0;
  for (int n = 0; n < NAn; ++n)
    if (lab[n] == 1) { fg++; if (fg > 128) lab[n] = -1; }
  int fgk = fg > 128 ? 128 : fg;
  int nbg = 256 - fgk, bg = 0;
  for (int n = 0; n < NAn; ++n)
    if (lab[n] == 0) { bg++; if (bg > nbg) lab[n] = -1; }
  int nex = 0;
  for (int n = 0; n < NAn; ++n) if (lab[n] >= 0) nex++;
  NEX[b] = (float)(nex < 1 ? 1 : nex);
}

__global__ void tgt_k(const float* __restrict__ gt, const int* __restrict__ AMX,
                      float* __restrict__ TGT) {
  int i = blockIdx.x * 256 + threadIdx.x;
  if (i >= Bc * NAn) return;
  int b = i / NAn, n = i - b * NAn;
  float x1, y1, x2, y2; anchor_of(n, x1, y1, x2, y2);
  const float* gp = gt + ((size_t)b * Gc + AMX[i]) * 5;
  float ew = x2 - x1 + 1.0f, eh = y2 - y1 + 1.0f;
  float ecx = x1 + 0.5f * ew, ecy = y1 + 0.5f * eh;
  float gw = gp[2] - gp[0] + 1.0f, gh = gp[3] - gp[1] + 1.0f;
  float gcx = gp[0] + 0.5f * gw, gcy = gp[1] + 0.5f * gh;
  float* t = TGT + (size_t)i * 4;
  t[0] = (gcx - ecx) / ew; t[1] = (gcy - ecy) / eh;
  t[2] = logf(gw / ew);    t[3] = logf(gh / eh);
}

__global__ void cls_loss_k(const float* __restrict__ CLS, const int* __restrict__ LAB,
                           float* __restrict__ acc) {
  int i = blockIdx.x * 256 + threadIdx.x;
  if (i >= Bc * NAn) return;
  int lab = LAB[i];
  if (lab == -1) return;
  int b = i / NAn, n = i - b * NAn;
  int m = n / ANC, a = n - m * ANC;
  float c0 = CLS[((size_t)b * MpD + m) * 18 + a];
  float c1 = CLS[((size_t)b * MpD + m) * 18 + 9 + a];
  float mx = fmaxf(c0, c1);
  float lse = mx + logf(expf(c0 - mx) + expf(c1 - mx));
  float pick = ((lab == 0) ? c0 : c1) - lse;
  atomicAdd(&acc[0], -pick);
  atomicAdd(&acc[1], 1.0f);
}

__global__ void reg_loss_k(const float* __restrict__ REGo, const int* __restrict__ LAB,
                           const float* __restrict__ TGT, const float* __restrict__ NEX,
                           float* __restrict__ acc) {
  int i = blockIdx.x * 256 + threadIdx.x;
  if (i >= Bc * NAn) return;
  if (LAB[i] != 1) return;  // only fg rows contribute (inw=0 elsewhere -> l=0)
  int b = i / NAn, n = i - b * NAn;
  int m = n / ANC, a = n - m * ANC;
  const float* pr = REGo + ((size_t)b * MpD + m) * 36 + a * 4;
  const float* tg = TGT + (size_t)i * 4;
  float s = 0.0f;
  for (int d = 0; d < 4; ++d) {
    float dd = pr[d] - tg[d];
    float ad = fabsf(dd);
    s += (ad < (1.0f / 9.0f)) ? 4.5f * dd * dd : ad - (0.5f / 9.0f);
  }
  atomicAdd(&acc[2], s / NEX[b]);
}

__global__ void finalize_k(const float* __restrict__ acc, float* __restrict__ out) {
  if (threadIdx.x == 0) {
    out[Bc * POST * 5 + 0] = acc[0] / fmaxf(acc[1], 1.0f);
    out[Bc * POST * 5 + 1] = acc[2] / (float)Bc;
  }
}

// ---------------- launcher ----------------
extern "C" void kernel_launch(void* const* d_in, const int* in_sizes, int n_in,
                              void* d_out, int out_size, void* d_ws, size_t ws_size,
                              hipStream_t stream) {
  (void)in_sizes; (void)n_in; (void)out_size; (void)ws_size;
  const float* feature = (const float*)d_in[0];
  const float* gt      = (const float*)d_in[1];
  const float* iminfo  = (const float*)d_in[2];
  const float* w_conv  = (const float*)d_in[3];
  const float* b_conv  = (const float*)d_in[4];
  const float* w_cls   = (const float*)d_in[5];
  const float* b_cls   = (const float*)d_in[6];
  const float* w_reg   = (const float*)d_in[7];
  const float* b_reg   = (const float*)d_in[8];
  float* out = (float*)d_out;
  char* ws = (char*)d_ws;

  _Float16* Wh   = (_Float16*)(ws + OFF_WH);
  _Float16* Fh   = (_Float16*)(ws + OFF_FH);
  _Float16* Xh   = (_Float16*)(ws + OFF_XH);
  _Float16* Wcr  = (_Float16*)(ws + OFF_WCR);
  float*    CLS  = (float*)(ws + OFF_CLS);
  float*    REGo = (float*)(ws + OFF_REG);
  float*    BOX  = (float*)(ws + OFF_BOX);
  unsigned long long* KEY = (unsigned long long*)(ws + OFF_KEY);
  float*    TB   = (float*)(ws + OFF_TB);
  int*      LAB  = (int*)(ws + OFF_LAB);
  int*      AMX  = (int*)(ws + OFF_AMX);
  float*    MOV  = (float*)(ws + OFF_MOV);
  float*    GTM  = (float*)(ws + OFF_GTM);
  float*    TGT  = (float*)(ws + OFF_TGT);
  float*    NEX  = (float*)(ws + OFF_NEX);
  float*    ACC  = (float*)(ws + OFF_ACC);

  auto nb = [](long n) { return (int)((n + 255) / 256); };

  zero_acc_k<<<1, 64, 0, stream>>>(ACC);
  pack_w_k  <<<nb((long)Kc * COc),      256, 0, stream>>>(w_conv, Wh);
  pack_f_k  <<<nb((long)Bc * Hp * Wp * Cc), 256, 0, stream>>>(feature, Fh);
  pack_wcr_k<<<nb((long)COc * 64),      256, 0, stream>>>(w_cls, w_reg, Wcr);

  conv3_wmma_k<<<dim3(30, 16, Bc), 256, 0, stream>>>(Fh, Wh, b_conv, Xh);
  conv1_wmma_k<<<dim3(60, 1, Bc),  256, 0, stream>>>(Xh, Wcr, b_cls, b_reg, CLS, REGo);

  proposal_prep_k<<<nb((long)Bc * LEN), 256, 0, stream>>>(CLS, REGo, iminfo, BOX, KEY);
  for (int k = 2; k <= LEN; k <<= 1)
    for (int j = k >> 1; j > 0; j >>= 1)
      bitonic_pass_k<<<dim3(LEN / 256, Bc), 256, 0, stream>>>(KEY, k, j);
  gather_top_k<<<nb((long)Bc * PRE), 256, 0, stream>>>(KEY, BOX, TB);
  nms_k<<<Bc, 1024, PRE * 16 + PRE * 4, stream>>>(TB, out);

  maxov_k <<<nb((long)Bc * NAn), 256, 0, stream>>>(gt, iminfo, MOV, AMX);
  gtmax_k <<<1, 64, 0, stream>>>(gt, iminfo, GTM);
  labels_k<<<nb((long)Bc * NAn), 256, 0, stream>>>(gt, iminfo, MOV, GTM, LAB);
  sample_k<<<1, 32, 0, stream>>>(LAB, NEX);
  tgt_k   <<<nb((long)Bc * NAn), 256, 0, stream>>>(gt, AMX, TGT);
  cls_loss_k<<<nb((long)Bc * NAn), 256, 0, stream>>>(CLS, LAB, ACC);
  reg_loss_k<<<nb((long)Bc * NAn), 256, 0, stream>>>(REGo, LAB, TGT, NEX, ACC);
  finalize_k<<<1, 32, 0, stream>>>(ACC, out);
}
